// GCDAD_12292196401229
// MI455X (gfx1250) — compile-verified, matrix operand order checked by reading, not run on hardware
//
#include <hip/hip_runtime.h>
#include <math.h>

// ---------------------------------------------------------------------------
// Problem constants
// ---------------------------------------------------------------------------
#define BB   4
#define LL   105
#define MM   55          // channels
#define DD   256         // GRU hidden
#define GDIM 768         // 3*DD
#define HSTR 264         // padded LDS row stride (f16 elems) to avoid bank conflicts

typedef __attribute__((ext_vector_type(16))) _Float16 v16h;
typedef __attribute__((ext_vector_type(8)))  float    v8f;

// Output region offsets (floats) in d_out:
//   series  [3,B,L,D]     @ 0
//   prior   [3,B,L,D]     @ 322560
//   series1 [3,B,M,L,D]   @ 645120
//   prior1  [3,B,M,L,D]   @ 18385920
#define OFF_PRIOR    322560
#define OFF_SERIES1  645120
#define OFF_PRIOR1   18385920
#define STRIDE_I5D   5913600     // B*M*L*D
#define STRIDE_B5D   1478400     // M*L*D
#define STRIDE_M5D   26880       // L*D

// ---------------------------------------------------------------------------
// RevIN normalization: per (b,m) mean/var over L
// ---------------------------------------------------------------------------
__global__ void k_norm(const float* __restrict__ x, float* __restrict__ xn) {
    int tid = blockIdx.x * blockDim.x + threadIdx.x;
    if (tid >= BB * MM) return;
    int b = tid / MM, m = tid % MM;
    const float* px = x + b * LL * MM + m;
    float s = 0.f, s2 = 0.f;
    for (int l = 0; l < LL; ++l) { float v = px[l * MM]; s += v; s2 += v * v; }
    float mu  = s  * (1.0f / LL);
    float var = s2 * (1.0f / LL) - mu * mu;
    float inv = rsqrtf(var + 1e-5f);
    float* pn = xn + b * LL * MM + m;
    for (int l = 0; l < LL; ++l) pn[l * MM] = (px[l * MM] - mu) * inv;
}

__global__ void k_zero(float* __restrict__ p, int n) {
    int i = blockIdx.x * blockDim.x + threadIdx.x;
    if (i < n) p[i] = 0.f;
}

// ---------------------------------------------------------------------------
// Input stream preparation kernels: build X[T][Npad][ins]
// ---------------------------------------------------------------------------
__global__ void k_prep1(float* __restrict__ X, const float* __restrict__ xn,
                        int p, int n, int Npad, int N) {
    int idx = blockIdx.x * blockDim.x + threadIdx.x;
    if (idx >= p * Npad) return;
    int t = idx / Npad, r = idx % Npad;
    float v = 0.f;
    if (r < N) {
        int b = r / (MM * n); int rem = r % (MM * n);
        int m = rem / n;      int j   = rem % n;
        int l = j * p + t;
        v = xn[(b * LL + l) * MM + m];
    }
    X[idx] = v;
}

__global__ void k_prep2(float* __restrict__ X, const float* __restrict__ xn,
                        int p, int n, int Npad, int N) {
    int idx = blockIdx.x * blockDim.x + threadIdx.x;
    if (idx >= n * Npad) return;
    int t = idx / Npad, r = idx % Npad;
    float v = 0.f;
    if (r < N) {
        int b = r / (MM * p); int rem = r % (MM * p);
        int m = rem / p;      int q   = rem % p;
        int l = q * n + t;
        v = xn[(b * LL + l) * MM + m];
    }
    X[idx] = v;
}

__global__ void k_prep3(float* __restrict__ X, const float* __restrict__ xn,
                        int Npad, int N) {
    int idx = blockIdx.x * blockDim.x + threadIdx.x;
    if (idx >= MM * Npad) return;
    int t = idx / Npad, r = idx % Npad;
    float v = 0.f;
    if (r < N) v = xn[r * MM + t];
    X[idx] = v;
}

__global__ void k_prep4(float* __restrict__ X, const float* __restrict__ xn,
                        int num, int front, int Npad, int N) {
    int idx = blockIdx.x * blockDim.x + threadIdx.x;
    if (idx >= MM * Npad * num) return;
    int t = idx / (Npad * num); int rem = idx % (Npad * num);
    int r = rem / num;          int k   = rem % num;
    float v = 0.f;
    if (r < N) {
        int b = r / LL, l = r % LL;
        int lc = l - front + k;
        lc = lc < 0 ? 0 : (lc > LL - 1 ? LL - 1 : lc);
        v = xn[(b * LL + lc) * MM + t];
    }
    X[idx] = v;
}

// ---------------------------------------------------------------------------
// WMMA GRU kernel (dynamic LDS):
//   layout: hl[16*HSTR] f16 | xl[55*16*INS] f32 | bl2[16*4096] f16
//   Gates r,z B-fragments register-resident; gate c B-fragments in per-wave
//   LDS slice (8KB/wave) to stay under 256 VGPRs (no scratch spill in loop).
//   blockIdx.y selects an encoder instance (enc3 merge); strides offset ptrs.
// mode 0: atomic accumulate into accA[(b*L + j*lmul + t)*D + d]
// mode 1: store h into outB[((b*M + t)*L + l)*D + d], r = b*L + l
// ---------------------------------------------------------------------------
template <int INS>
__global__ void __launch_bounds__(512)
k_gru(const float* __restrict__ X, int T, int N, int Npad,
      const float* __restrict__ w_ih, const float* __restrict__ w_hh,
      const float* __restrict__ b_ih, const float* __restrict__ b_hh,
      float* __restrict__ accA, float* __restrict__ outB,
      int mode, int sub, int lmul,
      int iStrWih, int iStrWhh, int iStrB, int iStrOut) {
    extern __shared__ char smem[];
    _Float16* hl  = (_Float16*)smem;                                   // 16*HSTR
    float*    xl  = (float*)(smem + 16 * HSTR * 2);                    // 55*16*INS
    _Float16* bl2 = (_Float16*)(smem + 16 * HSTR * 2 + MM * 16 * INS * 4); // 16*4096

    const int iy = blockIdx.y;
    w_ih += iy * iStrWih;  w_hh += iy * iStrWhh;
    b_ih += iy * iStrB;    b_hh += iy * iStrB;
    outB += (size_t)iy * iStrOut;

    const int tid  = threadIdx.x;
    const int lane = tid & 31;
    const int wv   = tid >> 5;        // hidden tile 0..15
    const int hf   = lane >> 4;       // lane half (K split)
    const int col  = lane & 15;       // N column within tile / A row
    const int dcol = wv * 16 + col;   // hidden column owned by this lane
    const int rowBase = blockIdx.x * 16;

    // --- preload w_hh B-fragments for gates r,z (f32 -> f16), VGPR-resident ---
    v16h bf0[8], bf1[8];
#pragma unroll
    for (int c = 0; c < 8; ++c) {
        const float* p0 = w_hh + (0 * DD + dcol) * DD + c * 32 + hf * 8;
        const float* p1 = w_hh + (1 * DD + dcol) * DD + c * 32 + hf * 8;
        v16h t0, t1;
#pragma unroll
        for (int e = 0; e < 8; ++e) {
            t0[e] = (_Float16)p0[e]; t0[e + 8] = (_Float16)p0[16 + e];
            t1[e] = (_Float16)p1[e]; t1[e + 8] = (_Float16)p1[16 + e];
        }
        bf0[c] = t0; bf1[c] = t1;
    }
    // --- gate c (candidate) B-fragments into per-wave LDS slice ---
#pragma unroll
    for (int c = 0; c < 8; ++c) {
        const float* p2 = w_hh + (2 * DD + dcol) * DD + c * 32 + hf * 8;
        _Float16* dst = bl2 + wv * 4096 + c * 512 + lane * 16;
#pragma unroll
        for (int e = 0; e < 8; ++e) {
            dst[e] = (_Float16)p2[e]; dst[e + 8] = (_Float16)p2[16 + e];
        }
    }

    // --- per-lane input-weight / bias constants for column dcol ---
    float wih0[INS], wih1[INS], wih2[INS];
#pragma unroll
    for (int k = 0; k < INS; ++k) {
        wih0[k] = w_ih[(0 * DD + dcol) * INS + k];
        wih1[k] = w_ih[(1 * DD + dcol) * INS + k];
        wih2[k] = w_ih[(2 * DD + dcol) * INS + k];
    }
    const float bi0 = b_ih[dcol], bi1 = b_ih[DD + dcol], bi2 = b_ih[2 * DD + dcol];
    const float bh0 = b_hh[dcol], bh1 = b_hh[DD + dcol], bh2 = b_hh[2 * DD + dcol];

    // --- stage this tile's input stream into LDS ---
    for (int idx = tid; idx < T * 16 * INS; idx += 512) {
        int t = idx / (16 * INS); int rem = idx % (16 * INS);
        int row = rem / INS;      int k   = rem % INS;
        xl[idx] = X[(t * Npad + rowBase + row) * INS + k];
    }
    // --- zero h tile ---
    for (int idx = tid; idx < 16 * HSTR; idx += 512) hl[idx] = (_Float16)0.f;
    __syncthreads();

    float hprev[8];
#pragma unroll
    for (int v = 0; v < 8; ++v) hprev[v] = 0.f;

    for (int t = 0; t < T; ++t) {
        v8f a0, a1, a2;
#pragma unroll
        for (int v = 0; v < 8; ++v) { a0[v] = bh0; a1[v] = bh1; a2[v] = bh2; }

#pragma unroll
        for (int c = 0; c < 8; ++c) {
            const _Float16* rp = hl + col * HSTR + c * 32 + hf * 8;
            v16h af;
#pragma unroll
            for (int e = 0; e < 8; ++e) { af[e] = rp[e]; af[e + 8] = rp[16 + e]; }
            const _Float16* b2p = bl2 + wv * 4096 + c * 512 + lane * 16;
            v16h b2;
#pragma unroll
            for (int e = 0; e < 16; ++e) b2[e] = b2p[e];
            a0 = __builtin_amdgcn_wmma_f32_16x16x32_f16(false, af, false, bf0[c],
                                                        (short)0, a0, false, false);
            a1 = __builtin_amdgcn_wmma_f32_16x16x32_f16(false, af, false, bf1[c],
                                                        (short)0, a1, false, false);
            a2 = __builtin_amdgcn_wmma_f32_16x16x32_f16(false, af, false, b2,
                                                        (short)0, a2, false, false);
        }
        __syncthreads();   // everyone done reading h(t-1)

#pragma unroll
        for (int v = 0; v < 8; ++v) {
            int row = v + hf * 8;            // C layout: M = v (lanes 0-15) / v+8
            float g0 = bi0, g1 = bi1, g2 = bi2;
#pragma unroll
            for (int k = 0; k < INS; ++k) {
                float xv = xl[(t * 16 + row) * INS + k];
                g0 += xv * wih0[k]; g1 += xv * wih1[k]; g2 += xv * wih2[k];
            }
            float rr = 1.f / (1.f + __expf(-(g0 + a0[v])));
            float zz = 1.f / (1.f + __expf(-(g1 + a1[v])));
            float cc = tanhf(g2 + rr * a2[v]);
            float hn = (1.f - zz) * cc + zz * hprev[v];
            hprev[v] = hn;
            hl[row * HSTR + dcol] = (_Float16)hn;

            int gr = rowBase + row;
            if (gr < N) {
                if (mode == 0) {
                    int b = gr / (MM * sub);
                    int j = gr % sub;
                    int l = j * lmul + t;
                    atomicAdd(accA + (b * LL + l) * DD + dcol, hn);
                } else {
                    int b = gr / LL, l = gr % LL;
                    outB[((b * MM + t) * LL + l) * DD + dcol] = hn;
                }
            }
        }
        __syncthreads();   // h(t) visible before next step's A loads
    }
}

// dynamic LDS bytes per instantiation
static inline int gru_smem_bytes(int ins) {
    return 16 * HSTR * 2 + MM * 16 * ins * 4 + 16 * 4096 * 2;
}

// ---------------------------------------------------------------------------
// mean over channels (/M) + softmax over D (contiguous 256) for series/prior
// ---------------------------------------------------------------------------
__global__ void k_softD(const float* __restrict__ acc, float* __restrict__ out) {
    __shared__ float sm[256];
    int row = blockIdx.x, tid = threadIdx.x;
    float v = acc[row * DD + tid] * (1.0f / MM);
    sm[tid] = v; __syncthreads();
    for (int s = 128; s > 0; s >>= 1) {
        if (tid < s) sm[tid] = fmaxf(sm[tid], sm[tid + s]);
        __syncthreads();
    }
    float mx = sm[0]; __syncthreads();
    float e = __expf(v - mx);
    sm[tid] = e; __syncthreads();
    for (int s = 128; s > 0; s >>= 1) {
        if (tid < s) sm[tid] += sm[tid + s];
        __syncthreads();
    }
    out[row * DD + tid] = e / sm[0];
}

// ---------------------------------------------------------------------------
// in-place softmax over M (stride L*D) for series1/prior1 regions
// ---------------------------------------------------------------------------
__global__ void k_softM(float* __restrict__ out) {
    int idx = blockIdx.x * blockDim.x + threadIdx.x;
    if (idx >= 2 * 3 * BB * LL * DD) return;
    int region = idx / (3 * BB * LL * DD);
    int rem    = idx % (3 * BB * LL * DD);
    int i    = rem / (BB * LL * DD);
    int rem2 = rem % (BB * LL * DD);
    int b    = rem2 / (LL * DD);
    int rem3 = rem2 % (LL * DD);   // l*D + d
    float* p = out + (region ? OFF_PRIOR1 : OFF_SERIES1)
                   + i * STRIDE_I5D + b * STRIDE_B5D + rem3;
    float v[MM];
    float mx = -1e30f;
#pragma unroll 1
    for (int m = 0; m < MM; ++m) { v[m] = p[m * STRIDE_M5D]; mx = fmaxf(mx, v[m]); }
    float s = 0.f;
#pragma unroll 1
    for (int m = 0; m < MM; ++m) { v[m] = __expf(v[m] - mx); s += v[m]; }
    float inv = 1.f / s;
#pragma unroll 1
    for (int m = 0; m < MM; ++m) p[m * STRIDE_M5D] = v[m] * inv;
}

// ---------------------------------------------------------------------------
// Host launcher
// ---------------------------------------------------------------------------
extern "C" void kernel_launch(void* const* d_in, const int* in_sizes, int n_in,
                              void* d_out, int out_size, void* d_ws, size_t ws_size,
                              hipStream_t stream) {
    (void)in_sizes; (void)n_in; (void)out_size; (void)ws_size;
    const float* x     = (const float*)d_in[0];
    const float* wih1  = (const float*)d_in[1];
    const float* whh1  = (const float*)d_in[2];
    const float* bih1  = (const float*)d_in[3];
    const float* bhh1  = (const float*)d_in[4];
    const float* wih2  = (const float*)d_in[5];
    const float* whh2  = (const float*)d_in[6];
    const float* bih2  = (const float*)d_in[7];
    const float* bhh2  = (const float*)d_in[8];
    const float* wih3  = (const float*)d_in[9];    // [3,768,1]
    const float* whh3  = (const float*)d_in[10];   // [3,768,256]
    const float* bih3  = (const float*)d_in[11];   // [3,768]
    const float* bhh3  = (const float*)d_in[12];
    float* out = (float*)d_out;

    // workspace layout (floats)
    float* ws  = (float*)d_ws;
    float* xn  = ws;                  // 23104
    float* acc = ws + 23104;          // 645120  ([2][3][B][L][D])
    float* X3  = ws + 668224;         // 23760
    float* X1b = ws + 692000;         // 3 * 24000
    float* X2b = ws + 764000;         // 3 * 24000
    float* X4b = ws + 836000;         // 3 * 167040

    const int P[3]   = {3, 5, 7};
    const int NN[3]  = {35, 21, 15};
    const int NUM[3] = {3, 5, 7};
    auto pad16 = [](int v) { return (v + 15) & ~15; };

    k_zero<<<(645120 + 255) / 256, 256, 0, stream>>>(acc, 645120);
    k_norm<<<1, 256, 0, stream>>>(x, xn);
    k_prep3<<<(MM * 432 + 255) / 256, 256, 0, stream>>>(X3, xn, 432, BB * LL);

    for (int i = 0; i < 3; ++i) {
        int p = P[i], n = NN[i], num = NUM[i], front = num / 2;
        int N1 = BB * MM * n, Np1 = pad16(N1);
        int N2 = BB * MM * p, Np2 = pad16(N2);
        k_prep1<<<(p * Np1 + 255) / 256, 256, 0, stream>>>(X1b + i * 24000, xn, p, n, Np1, N1);
        k_prep2<<<(n * Np2 + 255) / 256, 256, 0, stream>>>(X2b + i * 24000, xn, p, n, Np2, N2);
        k_prep4<<<(MM * 432 * num + 255) / 256, 256, 0, stream>>>(X4b + i * 167040, xn, num, front, 432, BB * LL);
    }

    const int sm1 = gru_smem_bytes(1);

    for (int i = 0; i < 3; ++i) {
        int p = P[i], n = NN[i];
        int N1 = BB * MM * n, Np1 = pad16(N1);
        int N2 = BB * MM * p, Np2 = pad16(N2);

        // enc1: T=p, accumulate -> series sums
        k_gru<1><<<dim3(Np1 / 16, 1), 512, sm1, stream>>>(
            X1b + i * 24000, p, N1, Np1, wih1, whh1, bih1, bhh1,
            acc + i * (BB * LL * DD), nullptr, 0, n, p, 0, 0, 0, 0);
        // enc2: T=n, accumulate -> prior sums
        k_gru<1><<<dim3(Np2 / 16, 1), 512, sm1, stream>>>(
            X2b + i * 24000, n, N2, Np2, wih2, whh2, bih2, bhh2,
            acc + (3 + i) * (BB * LL * DD), nullptr, 0, p, n, 0, 0, 0, 0);
    }

    // enc3: all three instances merged via gridDim.y (27 x 3 blocks)
    k_gru<1><<<dim3(432 / 16, 3), 512, sm1, stream>>>(
        X3, MM, BB * LL, 432, wih3, whh3, bih3, bhh3,
        nullptr, out + OFF_SERIES1, 1, 1, 1,
        GDIM, GDIM * DD, GDIM, STRIDE_I5D);

    // enc4: T=M, input size num, direct -> prior1 region
    for (int i = 0; i < 3; ++i) {
        int num = NUM[i];
        const float* wih4 = (const float*)d_in[13 + 4 * i];
        const float* whh4 = (const float*)d_in[14 + 4 * i];
        const float* bih4 = (const float*)d_in[15 + 4 * i];
        const float* bhh4 = (const float*)d_in[16 + 4 * i];
        float* o4 = out + OFF_PRIOR1 + i * STRIDE_I5D;
        switch (num) {
        case 3:
            k_gru<3><<<dim3(432 / 16, 1), 512, gru_smem_bytes(3), stream>>>(
                X4b + i * 167040, MM, BB * LL, 432,
                wih4, whh4, bih4, bhh4, nullptr, o4, 1, 1, 1, 0, 0, 0, 0);
            break;
        case 5:
            k_gru<5><<<dim3(432 / 16, 1), 512, gru_smem_bytes(5), stream>>>(
                X4b + i * 167040, MM, BB * LL, 432,
                wih4, whh4, bih4, bhh4, nullptr, o4, 1, 1, 1, 0, 0, 0, 0);
            break;
        default:
            k_gru<7><<<dim3(432 / 16, 1), 512, gru_smem_bytes(7), stream>>>(
                X4b + i * 167040, MM, BB * LL, 432,
                wih4, whh4, bih4, bhh4, nullptr, o4, 1, 1, 1, 0, 0, 0, 0);
            break;
        }
    }

    // series/prior: mean over M + softmax over D (2*3*B*L rows of 256)
    k_softD<<<2 * 3 * BB * LL, 256, 0, stream>>>(acc, out);
    // series1/prior1: in-place softmax over M
    k_softM<<<(2 * 3 * BB * LL * DD + 255) / 256, 256, 0, stream>>>(out);
}
